// NattenBlock_82489141887273
// MI455X (gfx1250) — matrix operand order repaired and checked by reading, not compiled, not run on hardware
//
#include <hip/hip_runtime.h>
#include <stdint.h>

typedef __bf16 bf16;
typedef bf16  v16bf __attribute__((ext_vector_type(16)));
typedef float v8f   __attribute__((ext_vector_type(8)));
typedef unsigned int u32x4 __attribute__((ext_vector_type(4)));
typedef int          i32x4 __attribute__((ext_vector_type(4)));
typedef int          i32x8 __attribute__((ext_vector_type(8)));

struct __align__(16) U4 { unsigned x, y, z, w; };
struct __align__(8)  U2 { unsigned x, y; };

union FragU { v16bf v; U4 u[2]; };
union Pack4 { bf16 h[4]; U2 u; };

// A-operand (16x32 bf16): lane m=lane&15, hi=lane>>4 holds K {hi*8..+7} and {hi*8+16..+23}
__device__ __forceinline__ v16bf frag_A(const bf16* base, int m, int rs, int c0) {
  FragU f;
  f.u[0] = *(const U4*)(base + m * rs + c0);
  f.u[1] = *(const U4*)(base + m * rs + c0 + 16);
  return f.v;
}
// B-operand (32x16 bf16) from [n][k] LDS image: lane n=lane&15, hi=lane>>4 holds K {hi*8..+7} and {hi*8+16..+23}
__device__ __forceinline__ v16bf frag_B(const bf16* base, int n, int rs, int c0) {
  FragU f;
  f.u[0] = *(const U4*)(base + n * rs + c0);
  f.u[1] = *(const U4*)(base + n * rs + c0 + 16);
  return f.v;
}

__device__ __forceinline__ v8f wmma_bf16(v16bf a, v16bf b, v8f c) {
  return __builtin_amdgcn_wmma_f32_16x16x32_bf16(false, a, false, b, (short)0, c, false, false);
}

#define NPIX   2304       // 48*48
#define DMODEL 768
#define PK     224        // padded key count (196 -> 14 x 16, multiple of 32)

// ---------------- Kernel 1: qkv = x @ w_qkv  (M=4608, N=2304, K=768) ----------------
__global__ __launch_bounds__(256) void qkv_gemm(const float* __restrict__ x,
                                                const float* __restrict__ w,
                                                bf16* __restrict__ q_ws,
                                                bf16* __restrict__ k_ws,
                                                bf16* __restrict__ v_ws) {
  __shared__ bf16 xs[128 * 32];   // [m][k]
  __shared__ bf16 bs[128 * 32];   // [n][k] (transposed image of W tile)
  const int tid  = threadIdx.x;
  const int lane = tid & 31;
  const int wave = tid >> 5;
  const int lidx = lane & 15, hi = lane >> 4;
  const int wm = wave >> 2, wn = wave & 3;           // 2x4 wave grid
  const int bm = blockIdx.x, bn = blockIdx.y;

  v8f acc[4][2];
  for (int mt = 0; mt < 4; ++mt)
    for (int nt = 0; nt < 2; ++nt)
      acc[mt][nt] = (v8f){0, 0, 0, 0, 0, 0, 0, 0};

  for (int kb = 0; kb < 24; ++kb) {
    // X tile: float4 loads, packed 4x bf16 LDS stores (b64)
    for (int idx = tid; idx < 128 * 8; idx += 256) {
      int m = idx >> 3, k4 = (idx & 7) * 4;
      float4 f = *(const float4*)(x + (size_t)(bm * 128 + m) * DMODEL + kb * 32 + k4);
      Pack4 p;
      p.h[0] = (bf16)f.x; p.h[1] = (bf16)f.y; p.h[2] = (bf16)f.z; p.h[3] = (bf16)f.w;
      *(U2*)(xs + m * 32 + k4) = p.u;
    }
    // W tile: coalesced float4 along N, transposed b16 stores into [n][k]
    for (int idx = tid; idx < 32 * 32; idx += 256) {
      int kk = idx >> 5, n4 = (idx & 31) * 4;
      float4 f = *(const float4*)(w + (size_t)(kb * 32 + kk) * 2304 + bn * 128 + n4);
      bs[(n4 + 0) * 32 + kk] = (bf16)f.x;
      bs[(n4 + 1) * 32 + kk] = (bf16)f.y;
      bs[(n4 + 2) * 32 + kk] = (bf16)f.z;
      bs[(n4 + 3) * 32 + kk] = (bf16)f.w;
    }
    if (kb + 1 < 24) {   // prefetch next K tile -> global_prefetch_b8
      __builtin_prefetch(x + (size_t)(bm * 128 + tid) * DMODEL + (kb + 1) * 32, 0, 1);
      __builtin_prefetch(w + (size_t)((kb + 1) * 32 + (tid & 31)) * 2304 + bn * 128, 0, 1);
    }
    __syncthreads();

    v16bf afr[4];
    for (int mt = 0; mt < 4; ++mt)
      afr[mt] = frag_A(xs + (wm * 64 + mt * 16) * 32, lidx, 32, hi * 8);
    for (int nt = 0; nt < 2; ++nt) {
      v16bf bfr = frag_B(bs + (wn * 32 + nt * 16) * 32, lidx, 32, hi * 8);
      for (int mt = 0; mt < 4; ++mt)
        acc[mt][nt] = wmma_bf16(afr[mt], bfr, acc[mt][nt]);
    }
    __syncthreads();
  }

  for (int mt = 0; mt < 4; ++mt)
    for (int nt = 0; nt < 2; ++nt) {
      int col = bn * 128 + wn * 32 + nt * 16 + lidx;
      int s = col / DMODEL, c7 = col - s * DMODEL;
      bf16* dst = (s == 0) ? q_ws : (s == 1 ? k_ws : v_ws);
      float scale = (s == 0) ? 0.125f : 1.0f;      // q * 64^-0.5
      for (int r = 0; r < 8; ++r) {
        int row = bm * 128 + wm * 64 + mt * 16 + r + hi * 8;
        dst[(size_t)row * DMODEL + c7] = (bf16)(acc[mt][nt][r] * scale);
      }
    }
}

// ---------------- Kernel 2: neighborhood attention, one block per (b, head, 8x8 tile) ----------------
__global__ __launch_bounds__(128) void natten_attn(const bf16* __restrict__ q_ws,
                                                   const bf16* __restrict__ k_ws,
                                                   const bf16* __restrict__ v_ws,
                                                   bf16* __restrict__ o_ws) {
  __shared__ bf16 kv_s[PK * 64];   // phase 1: K patch [key][64]; phase 2: V [dim][PK]
  __shared__ bf16 s_s[64 * PK];    // scores -> P (bf16)

  const int bid  = blockIdx.x;
  const int tile = bid % 36;
  const int tmp  = bid / 36;
  const int head = tmp % 12;
  const int b    = tmp / 12;
  const int ty = tile / 6, tx = tile % 6;
  const int py0 = min(max(ty * 8 - 3, 0), 34);
  const int px0 = min(max(tx * 8 - 3, 0), 34);

  const int tid  = threadIdx.x;
  const int lane = tid & 31;
  const int wave = tid >> 5;         // = M-tile (4 waves -> 64 queries)
  const int lidx = lane & 15, hi = lane >> 4;
  const size_t baseimg = (size_t)b * NPIX;

  // ---- stage K patch [key][64] via the Tensor Data Mover ----
  // 3D tile: dim0 = 64 contiguous bf16 (one head slice), dim1 = 14 pixels
  // (stride DMODEL), dim2 = 14 image rows (stride 48*DMODEL). TDM writes the
  // tile linearly to LDS => exactly kv_s[key][64] with key = py*14+px.
  if (wave == 0) {
    uint64_t gaddr = (uint64_t)(uintptr_t)(k_ws +
        (baseimg + (size_t)py0 * 48 + px0) * DMODEL + head * 64);
    unsigned laddr = (unsigned)(uintptr_t)(&kv_s[0]);   // low 32 bits = LDS offset

    u32x4 g0;
    g0[0] = 1u;                                          // count=1 (valid, user mode)
    g0[1] = laddr;                                       // lds_addr
    g0[2] = (unsigned)(gaddr & 0xffffffffu);             // global_addr[31:0]
    g0[3] = (unsigned)((gaddr >> 32) & 0x1ffffffu) | (2u << 30);  // addr[56:32] | type=2

    i32x8 g1;
    g1[0] = (1 << 16);                    // data_size = 1 (2 bytes); no multicast/pad
    g1[1] = (64 << 16);                   // tensor_dim0[15:0] = 64
    g1[2] = (14 << 16);                   // tensor_dim0[31:16]=0 | tensor_dim1[15:0]=14
    g1[3] = (64 << 16);                   // tensor_dim1[31:16]=0 | tile_dim0 = 64
    g1[4] = 14 | (14 << 16);              // tile_dim1 = 14 | tile_dim2 = 14
    g1[5] = DMODEL;                       // tensor_dim0_stride[31:0] = 768
    g1[6] = (int)(36864u << 16);          // dim0_stride[47:32]=0 | dim1_stride[15:0]=48*768
    g1[7] = 0;                            // tensor_dim1_stride[47:16] = 0

    i32x4 g2;
    g2[0] = 14;                           // tensor_dim2
    g2[1] = 0; g2[2] = 0; g2[3] = 0;
    i32x4 g3 = (i32x4){0, 0, 0, 0};
    i32x8 g4 = (i32x8){0, 0, 0, 0, 0, 0, 0, 0};   // extra group (clang-23 6-arg form)

    __builtin_amdgcn_tensor_load_to_lds(g0, g1, g2, g3, g4, 0);
    __builtin_amdgcn_s_wait_tensorcnt(0);
  }
  __syncthreads();

  // ---- S = Q K^T : 64x224, Q A-fragments straight from global ----
  {
    int q  = wave * 16 + lidx;
    int qi = ty * 8 + (q >> 3), qj = tx * 8 + (q & 7);
    const bf16* qbase = q_ws + (baseimg + (size_t)qi * 48 + qj) * DMODEL + head * 64;
    for (int nt = 0; nt < 14; ++nt) {
      v8f acc = {0, 0, 0, 0, 0, 0, 0, 0};
      for (int kt = 0; kt < 2; ++kt) {
        FragU fa;
        fa.u[0] = *(const U4*)(qbase + kt * 32 + hi * 8);
        fa.u[1] = *(const U4*)(qbase + kt * 32 + hi * 8 + 16);
        v16bf bfr = frag_B(kv_s + (nt * 16) * 64, lidx, 64, kt * 32 + hi * 8);
        acc = wmma_bf16(fa.v, bfr, acc);
      }
      for (int r = 0; r < 8; ++r)
        s_s[(wave * 16 + r + hi * 8) * PK + nt * 16 + lidx] = (bf16)acc[r];
    }
  }
  __syncthreads();

  // ---- stage V transposed [dim][key], zero-padding keys >= 196 ----
  for (int idx = tid; idx < PK * 64; idx += 128) {
    int key = idx >> 6, e = idx & 63;
    bf16 val = (bf16)0.f;
    if (key < 196) {
      int py = key / 14, px = key - py * 14;
      size_t pix = baseimg + (size_t)(py0 + py) * 48 + (px0 + px);
      val = v_ws[pix * DMODEL + head * 64 + e];
    }
    kv_s[e * PK + key] = val;
  }

  // ---- masked softmax, one thread per query row ----
  if (tid < 64) {
    int qy = tid >> 3, qx = tid & 7;
    int i = ty * 8 + qy, j = tx * 8 + qx;
    int si = min(max(i - 3, 0), 41), sj = min(max(j - 3, 0), 41);
    int rs = si - py0, cs = sj - px0;   // window origin within 14x14 patch
    bf16* row = s_s + tid * PK;
    float mx = -3.0e38f;
    for (int a = 0; a < 7; ++a)
      for (int bb = 0; bb < 7; ++bb)
        mx = fmaxf(mx, (float)row[(rs + a) * 14 + (cs + bb)]);
    float sum = 0.f;
    for (int a = 0; a < 7; ++a)
      for (int bb = 0; bb < 7; ++bb)
        sum += __expf((float)row[(rs + a) * 14 + (cs + bb)] - mx);
    float inv = 1.0f / sum;
    for (int c = 0; c < PK; ++c) {
      int pa = c / 14, pb = c - pa * 14;
      bool valid = (c < 196) && pa >= rs && pa < rs + 7 && pb >= cs && pb < cs + 7;
      row[c] = valid ? (bf16)(__expf((float)row[c] - mx) * inv) : (bf16)0.f;
    }
  }
  __syncthreads();

  // ---- O = P @ V : 64x64, K = 224 ----
  for (int nt = 0; nt < 4; ++nt) {
    v8f acc = {0, 0, 0, 0, 0, 0, 0, 0};
    for (int kt = 0; kt < 7; ++kt) {
      v16bf a  = frag_A(s_s + (wave * 16) * PK, lidx, PK, kt * 32 + hi * 8);
      v16bf bv = frag_B(kv_s + (nt * 16) * PK, lidx, PK, kt * 32 + hi * 8);
      acc = wmma_bf16(a, bv, acc);
    }
    for (int r = 0; r < 8; ++r) {
      int q  = wave * 16 + r + hi * 8;
      int qi = ty * 8 + (q >> 3), qj = tx * 8 + (q & 7);
      size_t pix = baseimg + (size_t)qi * 48 + qj;
      o_ws[pix * DMODEL + head * 64 + nt * 16 + lidx] = (bf16)acc[r];
    }
  }
}

// ---------------- Kernel 3: out = O @ w_out  (M=4608, N=768, K=768), fp32 out ----------------
__global__ __launch_bounds__(256) void out_gemm(const bf16* __restrict__ o_ws,
                                                const float* __restrict__ w,
                                                float* __restrict__ out) {
  __shared__ bf16 as[128 * 32];
  __shared__ bf16 bs[128 * 32];
  const int tid  = threadIdx.x;
  const int lane = tid & 31;
  const int wave = tid >> 5;
  const int lidx = lane & 15, hi = lane >> 4;
  const int wm = wave >> 2, wn = wave & 3;
  const int bm = blockIdx.x, bn = blockIdx.y;

  v8f acc[4][2];
  for (int mt = 0; mt < 4; ++mt)
    for (int nt = 0; nt < 2; ++nt)
      acc[mt][nt] = (v8f){0, 0, 0, 0, 0, 0, 0, 0};

  for (int kb = 0; kb < 24; ++kb) {
    // A tile already bf16: b128 copies
    for (int idx = tid; idx < 128 * 4; idx += 256) {
      int m = idx >> 2, k8 = (idx & 3) * 8;
      *(U4*)(as + m * 32 + k8) =
          *(const U4*)(o_ws + (size_t)(bm * 128 + m) * DMODEL + kb * 32 + k8);
    }
    // W tile: coalesced float4 along N, transposed stores
    for (int idx = tid; idx < 32 * 32; idx += 256) {
      int kk = idx >> 5, n4 = (idx & 31) * 4;
      float4 f = *(const float4*)(w + (size_t)(kb * 32 + kk) * DMODEL + bn * 128 + n4);
      bs[(n4 + 0) * 32 + kk] = (bf16)f.x;
      bs[(n4 + 1) * 32 + kk] = (bf16)f.y;
      bs[(n4 + 2) * 32 + kk] = (bf16)f.z;
      bs[(n4 + 3) * 32 + kk] = (bf16)f.w;
    }
    if (kb + 1 < 24)
      __builtin_prefetch(o_ws + (size_t)(bm * 128 + tid) * DMODEL + (kb + 1) * 32, 0, 1);
    __syncthreads();

    v16bf afr[4];
    for (int mt = 0; mt < 4; ++mt)
      afr[mt] = frag_A(as + (wm * 64 + mt * 16) * 32, lidx, 32, hi * 8);
    for (int nt = 0; nt < 2; ++nt) {
      v16bf bfr = frag_B(bs + (wn * 32 + nt * 16) * 32, lidx, 32, hi * 8);
      for (int mt = 0; mt < 4; ++mt)
        acc[mt][nt] = wmma_bf16(afr[mt], bfr, acc[mt][nt]);
    }
    __syncthreads();
  }

  for (int mt = 0; mt < 4; ++mt)
    for (int nt = 0; nt < 2; ++nt) {
      int col = bn * 128 + wn * 32 + nt * 16 + lidx;
      for (int r = 0; r < 8; ++r) {
        int row = bm * 128 + wm * 64 + mt * 16 + r + hi * 8;
        out[(size_t)row * DMODEL + col] = acc[mt][nt][r];
      }
    }
}

extern "C" void kernel_launch(void* const* d_in, const int* in_sizes, int n_in,
                              void* d_out, int out_size, void* d_ws, size_t ws_size,
                              hipStream_t stream) {
  const float* x     = (const float*)d_in[0];   // [2,48,48,768]
  const float* w_qkv = (const float*)d_in[1];   // [768,2304]
  const float* w_out = (const float*)d_in[2];   // [768,768]

  const size_t NELEM = (size_t)2 * NPIX * DMODEL;   // 4608*768 per tensor
  bf16* q_ws = (bf16*)d_ws;
  bf16* k_ws = q_ws + NELEM;
  bf16* v_ws = k_ws + NELEM;
  bf16* o_ws = v_ws + NELEM;

  dim3 g1(36, 18);
  qkv_gemm<<<g1, 256, 0, stream>>>(x, w_qkv, q_ws, k_ws, v_ws);

  natten_attn<<<2 * 12 * 36, 128, 0, stream>>>(q_ws, k_ws, v_ws, o_ws);

  dim3 g3(36, 6);
  out_gemm<<<g3, 256, 0, stream>>>(o_ws, w_out, (float*)d_out);
}
